// RelativeMultiheadAttention_27608049778899
// MI455X (gfx1250) — matrix operand order repaired and checked by reading, not compiled, not run on hardware
//
#include <hip/hip_runtime.h>
#include <hip/hip_bf16.h>
#include <hip/hip_fp16.h>

typedef __attribute__((ext_vector_type(16))) _Float16 v16h;
typedef __attribute__((ext_vector_type(8)))  _Float16 h8;
typedef __attribute__((ext_vector_type(8)))  float    v8f;
typedef __attribute__((ext_vector_type(4)))  float    f4;

#define AS_STRIDE 40   // halves: 32 K + 8 pad, keeps rows 16B aligned
#define BK 32

__device__ __forceinline__ h8 load8cvt(const float* p) {
  f4 a = *(const f4*)p;
  f4 b = *(const f4*)(p + 4);
  h8 r;
  r[0] = (_Float16)a[0]; r[1] = (_Float16)a[1]; r[2] = (_Float16)a[2]; r[3] = (_Float16)a[3];
  r[4] = (_Float16)b[0]; r[5] = (_Float16)b[1]; r[6] = (_Float16)b[2]; r[7] = (_Float16)b[3];
  return r;
}
__device__ __forceinline__ h8 load8cvt(const _Float16* p) { return *(const h8*)p; }

// C[M,N] = scale * (A @ op(B)) (+ bias), tiled 128 x BN, BK=32, 8 waves.
// BIAS: 0 none, 1 per-column vector, 2 relative-position bias table [4095,16] (head = blockIdx.z)
template<int BN, bool TRANSB, typename TA, typename TB, typename TC, int BIAS>
__global__ __launch_bounds__(256)
void wmma_gemm(const TA* __restrict__ A, long long aBatch, int lda,
               const TB* __restrict__ B, long long bBatch, int ldb,
               TC* __restrict__ C, long long cBatch, int ldc,
               const float* __restrict__ biasv,
               int M, int N, int K, float scale)
{
  constexpr int BM  = 128;
  constexpr int BSS = BN + 8;        // padded B stride in halves
  constexpr int WN  = BN / 4;        // wave column extent
  constexpr int MT  = 4;             // 64 rows / 16
  constexpr int NT  = WN / 16;

  __shared__ _Float16 As[2][BM * AS_STRIDE];
  __shared__ _Float16 Bs[2][BK * BSS];

  const int tid  = threadIdx.x;
  const int lane = tid & 31;
  const int wid  = tid >> 5;
  const int wrow = (wid >> 2) * 64;  // 2 row-waves
  const int wcol = (wid & 3) * WN;   // 4 col-waves
  const int z    = blockIdx.z;

  A += (long long)z * aBatch;
  B += (long long)z * bBatch;
  C += (long long)z * cBatch;

  const int m0 = blockIdx.y * BM;
  const int n0 = blockIdx.x * BN;

  v8f acc[MT][NT] = {};
  const int nk = K / BK;

  auto loadTile = [&](int kt, int buf) {
    const int k0 = kt * BK;
    // ---- A tile: 128 rows x 32 K, convert to f16 ----
    {
      int r  = tid >> 1;
      int cb = (tid & 1) * 16;
      const TA* src = A + (long long)(m0 + r) * lda + (k0 + cb);
      __builtin_prefetch(src + BK, 0, 0);     // global_prefetch_b8 for next stage
      h8 v0 = load8cvt(src);
      h8 v1 = load8cvt(src + 8);
      _Float16* dst = &As[buf][r * AS_STRIDE + cb];
      *(h8*)dst       = v0;
      *(h8*)(dst + 8) = v1;
    }
    // ---- B tile into Bs[k][n] layout ----
    if (TRANSB) {
      // B is [N,K]: read 32 contiguous K per n-row, scatter-transpose into LDS
      constexpr int TOT = BN * 2;  // chunks of 16 K
      #pragma unroll
      for (int i = 0; i < (TOT + 255) / 256; ++i) {
        int gid = tid + i * 256;
        if ((TOT % 256 == 0) || gid < TOT) {
          int n  = gid >> 1;
          int kb = (gid & 1) * 16;
          const TB* src = B + (long long)(n0 + n) * ldb + (k0 + kb);
          __builtin_prefetch(src + BK, 0, 0);
          h8 v0 = load8cvt(src);
          h8 v1 = load8cvt(src + 8);
          _Float16* bs = &Bs[buf][0];
          #pragma unroll
          for (int j = 0; j < 8; ++j) bs[(kb + j)     * BSS + n] = v0[j];
          #pragma unroll
          for (int j = 0; j < 8; ++j) bs[(kb + 8 + j) * BSS + n] = v1[j];
        }
      }
    } else {
      // B is [K,N] row-major: straight copy/convert
      constexpr int CH  = BN / 8;
      constexpr int TOT = BK * CH;
      #pragma unroll
      for (int i = 0; i < (TOT + 255) / 256; ++i) {
        int gid = tid + i * 256;
        if ((TOT % 256 == 0) || gid < TOT) {
          int kr = gid / CH;
          int nb = (gid % CH) * 8;
          const TB* src = B + (long long)(k0 + kr) * ldb + (n0 + nb);
          h8 v = load8cvt(src);
          *(h8*)&Bs[buf][kr * BSS + nb] = v;
        }
      }
    }
  };

  loadTile(0, 0);
  __syncthreads();

  int buf = 0;
  for (int kt = 0; kt < nk; ++kt) {
    if (kt + 1 < nk) loadTile(kt + 1, buf ^ 1);

    // A fragment (16x32 f16): lane L<16 holds K {0..7,16..23}, L>=16 holds {8..15,24..31}
    v16h af[MT];
    #pragma unroll
    for (int mt = 0; mt < MT; ++mt) {
      const _Float16* pa = &As[buf][(wrow + mt * 16 + (lane & 15)) * AS_STRIDE + 8 * (lane >> 4)];
      h8 lo = *(const h8*)pa;
      h8 hi = *(const h8*)(pa + 16);
      af[mt] = __builtin_shufflevector(lo, hi, 0,1,2,3,4,5,6,7,8,9,10,11,12,13,14,15);
    }
    // B fragment (32x16 f16): lane L holds column N=L%16, K {0..7,16..23} (L<16) / {8..15,24..31} (L>=16)
    v16h bf[NT];
    #pragma unroll
    for (int nt = 0; nt < NT; ++nt) {
      const _Float16* pb = &Bs[buf][8 * (lane >> 4) * BSS + wcol + nt * 16 + (lane & 15)];
      #pragma unroll
      for (int s = 0; s < 16; ++s) bf[nt][s] = pb[((s & 7) + (s >> 3) * 16) * BSS];
    }
    #pragma unroll
    for (int mt = 0; mt < MT; ++mt)
      #pragma unroll
      for (int nt = 0; nt < NT; ++nt)
        acc[mt][nt] = __builtin_amdgcn_wmma_f32_16x16x32_f16(
            false, af[mt], false, bf[nt], (short)0, acc[mt][nt], false, false);

    __syncthreads();
    buf ^= 1;
  }

  // Epilogue: C/D layout -> VGPR r, lanes 0-15: M=r, N=lane; lanes 16-31: M=r+8, N=lane-16
  const int rbase = wrow + 8 * (lane >> 4);
  const int cl    = lane & 15;
  #pragma unroll
  for (int mt = 0; mt < MT; ++mt) {
    #pragma unroll
    for (int nt = 0; nt < NT; ++nt) {
      #pragma unroll
      for (int r = 0; r < 8; ++r) {
        int gm = m0 + rbase + mt * 16 + r;
        int gn = n0 + wcol + nt * 16 + cl;
        float v = acc[mt][nt][r] * scale;
        if (BIAS == 1) v += biasv[gn];
        if (BIAS == 2) v += biasv[(long long)(gn - gm + 2047) * 16 + z];
        C[(long long)gm * ldc + gn] = (TC)v;
      }
    }
  }
}

// In-place softmax over rows of 2048 fp32, one block per row.
__global__ __launch_bounds__(256)
void softmax_rows(float* __restrict__ attn, int ncols)
{
  float* p = attn + (long long)blockIdx.x * ncols;
  const int tid = threadIdx.x;
  float x[8];
  float m = -INFINITY;
  #pragma unroll
  for (int i = 0; i < 8; ++i) { x[i] = p[tid + i * 256]; m = fmaxf(m, x[i]); }

  __shared__ float red[256];
  red[tid] = m; __syncthreads();
  for (int s = 128; s > 0; s >>= 1) { if (tid < s) red[tid] = fmaxf(red[tid], red[tid + s]); __syncthreads(); }
  m = red[0]; __syncthreads();

  float sum = 0.f;
  #pragma unroll
  for (int i = 0; i < 8; ++i) { x[i] = __expf(x[i] - m); sum += x[i]; }
  red[tid] = sum; __syncthreads();
  for (int s = 128; s > 0; s >>= 1) { if (tid < s) red[tid] += red[tid + s]; __syncthreads(); }
  float inv = 1.0f / red[0];
  #pragma unroll
  for (int i = 0; i < 8; ++i) p[tid + i * 256] = x[i] * inv;
}

extern "C" void kernel_launch(void* const* d_in, const int* in_sizes, int n_in,
                              void* d_out, int out_size, void* d_ws, size_t ws_size,
                              hipStream_t stream) {
  const float* query   = (const float*)d_in[0];
  const float* key     = (const float*)d_in[1];
  const float* value   = (const float*)d_in[2];
  const float* wq_w    = (const float*)d_in[3];
  const float* wq_b    = (const float*)d_in[4];
  const float* wk_w    = (const float*)d_in[5];
  const float* wk_b    = (const float*)d_in[6];
  const float* wv_w    = (const float*)d_in[7];
  const float* wv_b    = (const float*)d_in[8];
  const float* dense_w = (const float*)d_in[9];
  const float* dense_b = (const float*)d_in[10];
  const float* relb    = (const float*)d_in[11];

  const int S = 2048, D = 1024, H = 16;
  float* out  = (float*)d_out;                         // [2048,1024]
  float* attn = (float*)d_out + (long long)S * D;      // [16,2048,2048]

  const long long NM = (long long)S * D;               // 2M elements
  _Float16* Qh  = (_Float16*)d_ws;                     // f16 intermediates in ws (16 MB total)
  _Float16* Kh  = Qh + NM;
  _Float16* Vh  = Kh + NM;
  _Float16* CTX = Vh + NM;

  dim3 blk(256);

  // 1) Projections: [2048,1024] @ W^T + b -> f16
  wmma_gemm<128, true, float, float, _Float16, 1><<<dim3(D/128, S/128, 1), blk, 0, stream>>>(
      query, 0LL, D, wq_w, 0LL, D, Qh, 0LL, D, wq_b, S, D, D, 1.0f);
  wmma_gemm<128, true, float, float, _Float16, 1><<<dim3(D/128, S/128, 1), blk, 0, stream>>>(
      key,   0LL, D, wk_w, 0LL, D, Kh, 0LL, D, wk_b, S, D, D, 1.0f);
  wmma_gemm<128, true, float, float, _Float16, 1><<<dim3(D/128, S/128, 1), blk, 0, stream>>>(
      value, 0LL, D, wv_w, 0LL, D, Vh, 0LL, D, wv_b, S, D, D, 1.0f);

  // 2) Scores per head: Qh @ Kh^T / 8 + rel_bias -> fp32 attn region
  wmma_gemm<128, true, _Float16, _Float16, float, 2><<<dim3(S/128, S/128, H), blk, 0, stream>>>(
      Qh, 64LL, D, Kh, 64LL, D, attn, (long long)S * S, S, relb, S, S, 64, 0.125f);

  // 3) Row softmax, in place
  softmax_rows<<<dim3(H * S), blk, 0, stream>>>(attn, S);

  // 4) ctx = attn @ Vh per head -> f16 ctx [2048,1024]
  wmma_gemm<64, false, float, _Float16, _Float16, 0><<<dim3(1, S/128, H), blk, 0, stream>>>(
      attn, (long long)S * S, S, Vh, 64LL, D, CTX, 64LL, D, nullptr, S, 64, S, 1.0f);

  // 5) out = ctx @ dense_w^T + dense_b -> fp32
  wmma_gemm<128, true, _Float16, float, float, 1><<<dim3(D/128, S/128, 1), blk, 0, stream>>>(
      CTX, 0LL, D, dense_w, 0LL, D, out, 0LL, D, dense_b, S, D, D, 1.0f);

  (void)in_sizes; (void)n_in; (void)out_size; (void)ws_size;
}
